// GeometryOnlyRegressor_16449724744479
// MI455X (gfx1250) — matrix operand order repaired and checked by reading, not compiled, not run on hardware
//
#include <hip/hip_runtime.h>
#include <hip/hip_bf16.h>

typedef __attribute__((ext_vector_type(16))) _Float16 v16h;
typedef __attribute__((ext_vector_type(8)))  float    v8f;

#define HID      192
#define NLAYER   5
#define LSTRIDE  196   // padded LDS row stride in floats (196 % 64 == 4 -> bank rotation)

#define WMMA_F16(Afrag, Bfrag, Cfrag) \
  __builtin_amdgcn_wmma_f32_16x16x32_f16(false, (Afrag), false, (Bfrag), (short)0, (Cfrag), false, false)

// ---------------------------------------------------------------------------
// Weight swizzle pre-pass: fp32 -> f16, arranged as WMMA B fragments.
//   B (32x16 f16) layout: lane<16 -> n=lane, k = {0..7, 16..23}
//                         lane>=16 -> n=lane-16, k = {8..15, 24..31}
//   => lane's 16 halves = column n of W^T, matching load_a_frag's k order.
// Tower: idx = ((l*12 + t)*6 + c)*512 + lane*16 + j ; value = W[l][n][k]
//        n = t*16 + (lane&15), k = c*32 + (lane>>4)*8 + (j&7) + (j>>3)*16
// ---------------------------------------------------------------------------
__global__ __launch_bounds__(256) void swizzle_tower(const float* __restrict__ W,
                                                     _Float16* __restrict__ out) {
  int idx = blockIdx.x * 256 + threadIdx.x;
  if (idx >= NLAYER * HID * HID) return;
  int l = idx / (HID * HID);
  int f = idx - l * HID * HID;
  int t = f / (6 * 512);
  int g = f - t * 6 * 512;
  int c = g >> 9;
  int q = g & 511;
  int lane = q >> 4, j = q & 15;
  int n = t * 16 + (lane & 15);
  int k = c * 32 + ((lane >> 4) << 3) + (j & 7) + ((j >> 3) << 4);
  out[idx] = (_Float16)W[(l * HID + n) * HID + k];
}

// head_W1: [192 out][576 in] -> frags [12 t][18 c][512]
__global__ __launch_bounds__(256) void swizzle_head(const float* __restrict__ W1,
                                                    _Float16* __restrict__ out) {
  int idx = blockIdx.x * 256 + threadIdx.x;
  if (idx >= 12 * 18 * 512) return;
  int t = idx / (18 * 512);
  int g = idx - t * 18 * 512;
  int c = g >> 9;
  int q = g & 511;
  int lane = q >> 4, j = q & 15;
  int n = t * 16 + (lane & 15);
  int k = c * 32 + ((lane >> 4) << 3) + (j & 7) + ((j >> 3) << 4);
  out[idx] = (_Float16)W1[n * 576 + k];
}

// Build a 16-bit A fragment (16x32 MxK) for one lane from an fp32 LDS row.
// lane<16 -> M=lane, K halves {k1..k1+7, k1+16..k1+23}; lane>=16 same M, +8.
__device__ __forceinline__ v16h load_a_frag(const float* __restrict__ rowp, int k1) {
  const float4* p1 = (const float4*)(rowp + k1);
  const float4* p2 = (const float4*)(rowp + k1 + 16);
  float4 a0 = p1[0], a1 = p1[1], b0 = p2[0], b1 = p2[1];
  v16h a;
  a[0]  = (_Float16)a0.x; a[1]  = (_Float16)a0.y; a[2]  = (_Float16)a0.z; a[3]  = (_Float16)a0.w;
  a[4]  = (_Float16)a1.x; a[5]  = (_Float16)a1.y; a[6]  = (_Float16)a1.z; a[7]  = (_Float16)a1.w;
  a[8]  = (_Float16)b0.x; a[9]  = (_Float16)b0.y; a[10] = (_Float16)b0.z; a[11] = (_Float16)b0.w;
  a[12] = (_Float16)b1.x; a[13] = (_Float16)b1.y; a[14] = (_Float16)b1.z; a[15] = (_Float16)b1.w;
  return a;
}

// ---------------------------------------------------------------------------
// Fused forward: block = 64 threads (2 waves), 16 molecules (32 rows) per
// block.  Row 2m = h1 of molecule m, row 2m+1 = h2.
// Waves split the 12 n-tiles (wave w -> tiles w*6..w*6+5); every wave covers
// all 32 rows as two A tiles, so each B fragment is loaded once per block and
// feeds two WMMAs (halves L2 weight traffic).
// ---------------------------------------------------------------------------
__global__ __launch_bounds__(64) void fused_forward(
    const int* __restrict__ z, const float* __restrict__ pos,
    const float* __restrict__ emb,
    const _Float16* __restrict__ Wfrag,     // [5][12][6][512] f16
    const float* __restrict__ layer_b, const float* __restrict__ layer_g,
    const float* __restrict__ layer_beta,
    const float* __restrict__ dist_W, const float* __restrict__ dist_b,
    const _Float16* __restrict__ W1frag,    // [12][18][512] f16
    const float* __restrict__ head_b1, const float* __restrict__ head_W2,
    const float* __restrict__ head_b2,
    float* __restrict__ out, int numMol) {
  __shared__ float hbuf[32][LSTRIDE];   // fp32 activations (residual stream)
  __shared__ float ybuf[32][LSTRIDE];   // fp32 pre-LN linear output
  __shared__ float dfeat[16][LSTRIDE];  // dist feature rows
  __shared__ float distv[16];
  __shared__ int   zrow[32];
  __shared__ float part[64][2];         // LN partial sums
  __shared__ float outacc[16];

  const int tid  = threadIdx.x;
  const int wave = tid >> 5;
  const int lane = tid & 31;
  const long molBase = (long)blockIdx.x * 16;

  // ---- phase 1: distances, z gather, accumulator init ----
  if (tid < 32) {
    long m = molBase + (tid >> 1);
    if (m >= numMol) m = numMol - 1;
    zrow[tid] = z[m * 8 + (tid & 1)];
  }
  if (tid < 16) {
    long m = molBase + tid;
    if (m >= numMol) m = numMol - 1;
    long a0 = m * 8, a1 = a0 + 1;   // batch = arange//8: first=8m, second=8m+1
    float dx = pos[a0 * 3 + 0] - pos[a1 * 3 + 0];
    float dy = pos[a0 * 3 + 1] - pos[a1 * 3 + 1];
    float dz = pos[a0 * 3 + 2] - pos[a1 * 3 + 2];
    distv[tid]  = sqrtf(dx * dx + dy * dy + dz * dz + 1e-12f);
    outacc[tid] = 0.0f;
  }
  __syncthreads();

  // ---- phase 2: embedding rows + dist feature rows (float4 vectorized) ----
  for (int idx = tid; idx < 32 * 48; idx += 64) {
    int r = idx / 48, k4 = idx - r * 48;
    const float4* src = (const float4*)(emb + (long)zrow[r] * HID);
    *(float4*)&hbuf[r][k4 * 4] = src[k4];
  }
  for (int idx = tid; idx < 16 * 48; idx += 64) {
    int m = idx / 48, k4 = idx - m * 48;
    float4 w  = ((const float4*)dist_W)[k4];
    float4 bb = ((const float4*)dist_b)[k4];
    float d = distv[m];
    float4 o; o.x = d * w.x + bb.x; o.y = d * w.y + bb.y;
    o.z = d * w.z + bb.z; o.w = d * w.w + bb.w;
    *(float4*)&dfeat[m][k4 * 4] = o;
  }
  __syncthreads();

  const int ksel  = (lane >> 4) << 3;   // A lane-group K shift (0 or 8)
  const int ncol  = lane & 15;          // C/D column within n-tile
  const int rowj  = (lane >> 4) << 3;   // C/D row offset (0 or 8)
  const int twave = wave * 6;           // this wave's first n-tile

  // ================= tower: 5 x (Linear + LN + ReLU + residual) ============
  for (int l = 0; l < NLAYER; ++l) {
    // single per-layer fragment base; all 72 loads use immediate offsets
    const v16h* bfL = (const v16h*)Wfrag + ((l * 12 + twave) * 6) * 32 + lane;
    if (l + 1 < NLAYER)  // warm next layer's fragments into near caches
      __builtin_prefetch((const void*)(bfL + 12 * 6 * 32), 0, 1);

    // A fragments for both 16-row tiles (built from LDS, fp32 -> f16)
    v16h A0[6], A1[6];
#pragma unroll
    for (int c = 0; c < 6; ++c) {
      A0[c] = load_a_frag(&hbuf[ncol][0],      c * 32 + ksel);
      A1[c] = load_a_frag(&hbuf[16 + ncol][0], c * 32 + ksel);
    }
    // per-tile biases, hoisted out of the hot loop
    float bias6[6];
#pragma unroll
    for (int tt = 0; tt < 6; ++tt)
      bias6[tt] = layer_b[l * HID + (twave + tt) * 16 + ncol];

#pragma unroll
    for (int tt = 0; tt < 6; ++tt) {
      const int t = twave + tt;
      const float b0 = bias6[tt];
      v8f a0 = {b0, b0, b0, b0, b0, b0, b0, b0};
      v8f a1 = a0;
      v16h Bc = bfL[tt * 6 * 32];         // double-buffered B fragments
#pragma unroll
      for (int c = 0; c < 6; ++c) {
        v16h Bn = Bc;
        if (c < 5) Bn = bfL[(tt * 6 + c + 1) * 32];
        a0 = WMMA_F16(A0[c], Bc, a0);     // two independent acc chains
        a1 = WMMA_F16(A1[c], Bc, a1);
        Bc = Bn;
      }
#pragma unroll
      for (int j = 0; j < 8; ++j) {
        ybuf[rowj + j][t * 16 + ncol]      = a0[j];
        ybuf[16 + rowj + j][t * 16 + ncol] = a1[j];
      }
    }
    __syncthreads();

    // LayerNorm(192) + ReLU + residual: thread (r = tid&31, half = tid>>5)
    {
      const int r  = tid & 31;
      const int k0 = (tid >> 5) * 96;
      float s = 0.f, s2 = 0.f;
#pragma unroll 4
      for (int k = 0; k < 24; ++k) {
        float4 v = *(const float4*)&ybuf[r][k0 + k * 4];
        s  += v.x + v.y + v.z + v.w;
        s2 += v.x * v.x + v.y * v.y + v.z * v.z + v.w * v.w;
      }
      part[tid][0] = s; part[tid][1] = s2;
      __syncthreads();
      s  += part[tid ^ 32][0];
      s2 += part[tid ^ 32][1];
      float mu  = s  * (1.0f / HID);
      float var = s2 * (1.0f / HID) - mu * mu;
      float inv = rsqrtf(var + 1e-5f);
#pragma unroll 4
      for (int k = 0; k < 24; ++k) {
        const int kk = k0 + k * 4;
        float4 y  = *(const float4*)&ybuf[r][kk];
        float4 g4 = *(const float4*)(layer_g    + l * HID + kk);
        float4 be = *(const float4*)(layer_beta + l * HID + kk);
        float4 h  = *(float4*)&hbuf[r][kk];
        h.x += fmaxf((y.x - mu) * inv * g4.x + be.x, 0.f);
        h.y += fmaxf((y.y - mu) * inv * g4.y + be.y, 0.f);
        h.z += fmaxf((y.z - mu) * inv * g4.z + be.z, 0.f);
        h.w += fmaxf((y.w - mu) * inv * g4.w + be.w, 0.f);
        *(float4*)&hbuf[r][kk] = h;
      }
    }
    __syncthreads();
  }

  // ================= head: relu([h1|h2|df] @ W1^T + b1) @ W2^T + b2 ========
  // A rows = 16 molecules; waves split the 12 n-tiles; K = 576 in 3 segments.
  const v16h* bfH = (const v16h*)W1frag + (twave * 18) * 32 + lane;
  __builtin_prefetch((const void*)bfH, 0, 1);

  v8f hacc[6];                       // persistent accumulators (one per n-tile)
#pragma unroll
  for (int tt = 0; tt < 6; ++tt) {
    float b0 = head_b1[(twave + tt) * 16 + ncol];
    v8f a = {b0, b0, b0, b0, b0, b0, b0, b0};
    hacc[tt] = a;
  }

#pragma unroll
  for (int seg = 0; seg < 3; ++seg) {
    // A fragments for this K segment, built once, reused by all 6 n-tiles
    const float* rowp = (seg == 0) ? &hbuf[2 * ncol][0]
                      : (seg == 1) ? &hbuf[2 * ncol + 1][0]
                                   : &dfeat[ncol][0];
    v16h A6[6];
#pragma unroll
    for (int c = 0; c < 6; ++c)
      A6[c] = load_a_frag(rowp, c * 32 + ksel);

#pragma unroll
    for (int tt = 0; tt < 6; ++tt) {
      v8f a = hacc[tt];
      v16h Bc = bfH[(tt * 18 + seg * 6) * 32];
#pragma unroll
      for (int c = 0; c < 6; ++c) {
        v16h Bn = Bc;
        if (c < 5) Bn = bfH[(tt * 18 + seg * 6 + c + 1) * 32];
        a = WMMA_F16(A6[c], Bc, a);
        Bc = Bn;
      }
      hacc[tt] = a;
    }
  }

  // fused ReLU * W2 partial dot, reduced through LDS float atomics
  float racc[8];
#pragma unroll
  for (int j = 0; j < 8; ++j) racc[j] = 0.f;
#pragma unroll
  for (int tt = 0; tt < 6; ++tt) {
    float w2v = head_W2[(twave + tt) * 16 + ncol];
#pragma unroll
    for (int j = 0; j < 8; ++j)
      racc[j] += fmaxf(hacc[tt][j], 0.f) * w2v;
  }
#pragma unroll
  for (int j = 0; j < 8; ++j)
    atomicAdd(&outacc[rowj + j], racc[j]);   // ds_add_f32
  __syncthreads();

  if (tid < 16) {
    long m = molBase + tid;
    if (m < numMol) out[m] = outacc[tid] + head_b2[0];
  }
}

// ---------------------------------------------------------------------------
extern "C" void kernel_launch(void* const* d_in, const int* in_sizes, int n_in,
                              void* d_out, int out_size, void* d_ws, size_t ws_size,
                              hipStream_t stream) {
  // setup_inputs() order:
  // 0 images, 1 z, 2 pos, 3 batch, 4 num_molecules, 5 emb, 6 layer_W,
  // 7 layer_b, 8 layer_g, 9 layer_beta, 10 dist_W, 11 dist_b,
  // 12 head_W1, 13 head_b1, 14 head_W2, 15 head_b2
  const int*   z        = (const int*)  d_in[1];
  const float* pos      = (const float*)d_in[2];
  const float* emb      = (const float*)d_in[5];
  const float* layerW   = (const float*)d_in[6];
  const float* layerB   = (const float*)d_in[7];
  const float* layerG   = (const float*)d_in[8];
  const float* layerBe  = (const float*)d_in[9];
  const float* distW    = (const float*)d_in[10];
  const float* distB    = (const float*)d_in[11];
  const float* headW1   = (const float*)d_in[12];
  const float* headB1   = (const float*)d_in[13];
  const float* headW2   = (const float*)d_in[14];
  const float* headB2   = (const float*)d_in[15];
  float* out = (float*)d_out;

  const int numMol = in_sizes[1] / 8;   // N = B * 8 atoms

  _Float16* wsW  = (_Float16*)d_ws;                  // 5*192*192 halves
  _Float16* wsW1 = wsW + NLAYER * HID * HID;         // 12*18*512 halves

  const int nTower = NLAYER * HID * HID;
  swizzle_tower<<<(nTower + 255) / 256, 256, 0, stream>>>(layerW, wsW);
  const int nHead = 12 * 18 * 512;
  swizzle_head<<<(nHead + 255) / 256, 256, 0, stream>>>(headW1, wsW1);

  const int nBlocks = (numMol + 15) / 16;
  fused_forward<<<nBlocks, 64, 0, stream>>>(z, pos, emb, wsW, layerB, layerG,
                                            layerBe, distW, distB, wsW1,
                                            headB1, headW2, headB2, out, numMol);
}
